// GINC_83846351552684
// MI455X (gfx1250) — matrix operand order repaired and checked by reading, not compiled, not run on hardware
//
#include <hip/hip_runtime.h>
#include <hip/hip_bf16.h>

typedef __attribute__((ext_vector_type(16))) _Float16 v16h;
typedef __attribute__((ext_vector_type(8)))  float    v8f;

#define N_NODES_C 100000
#define N_EDGES_C 1600000
#define DIM       128
#define NGRAPH    64
#define POOLDIM   256

// ---------------- workspace layout (bytes, 256-aligned) ----------------
static const size_t FA_OFF   = 0;           // N*128 f32  aggregation buffer      (51,200,000)
static const size_t FB_OFF   = 51200000;    // N*128 f32  feature buffer          (51,200,000)
static const size_t TMP_OFF  = 102400000;   // N*128 f16  inner-MLP activations   (25,600,000)
static const size_t WH_OFF   = 128000000;   // 6 * 128*128 f16 packed weight frags  (196,608)
static const size_t POOL_OFF = 128196608;   // 64*128 f32 pool + 64 f32 cnt          (33,024)
static const size_t OFFS_OFF = 128229632;   // (N+1) i32 CSR row offsets            (400,128)
static const size_t CURS_OFF = 128629760;   // N i32 CSR cursors / degrees          (400,128)
static const size_t SRC_OFF  = 129029888;   // E i32 CSR source indices           (6,400,000)
static const size_t BSUM_OFF = 135429888;   // 512 i32 scan block sums                (2,048)
// total ~135.4 MB

// ---------------- generic zero ----------------
__global__ void zero_u32_kernel(unsigned int* __restrict__ p, int n) {
    int i = blockIdx.x * blockDim.x + threadIdx.x;
    if (i < n) p[i] = 0u;
}

// ================= CSR build (once per launch; reused by all 3 layers) =================
__global__ void degree_kernel(const long long* __restrict__ ei, int* __restrict__ deg) {
    int e = blockIdx.x * blockDim.x + threadIdx.x;
    if (e >= N_EDGES_C) return;
    atomicAdd(&deg[(int)ei[(long long)N_EDGES_C + e]], 1);
}

// per-block exclusive scan of degrees + block totals
__global__ void scan_partial_kernel(const int* __restrict__ deg, int* __restrict__ offs,
                                    int* __restrict__ bsum) {
    __shared__ int tmp[256];
    const int tid = threadIdx.x;
    const int i   = blockIdx.x * 256 + tid;
    const int v   = (i < N_NODES_C) ? deg[i] : 0;
    tmp[tid] = v;
    __syncthreads();
    #pragma unroll
    for (int d = 1; d < 256; d <<= 1) {     // Hillis-Steele inclusive scan
        int t = (tid >= d) ? tmp[tid - d] : 0;
        __syncthreads();
        tmp[tid] += t;
        __syncthreads();
    }
    if (i < N_NODES_C) offs[i] = tmp[tid] - v;          // exclusive within block
    if (tid == 255) bsum[blockIdx.x] = tmp[255];
}

__global__ void scan_bsum_kernel(int* __restrict__ bsum, int nb) {
    if (blockIdx.x == 0 && threadIdx.x == 0) {
        int run = 0;
        for (int b = 0; b < nb; ++b) { int v = bsum[b]; bsum[b] = run; run += v; }
    }
}

__global__ void scan_addback_kernel(int* __restrict__ offs, const int* __restrict__ bsum) {
    int i = blockIdx.x * 256 + threadIdx.x;
    if (i < N_NODES_C) offs[i] += bsum[blockIdx.x];
    if (i == 0) offs[N_NODES_C] = N_EDGES_C;
}

__global__ void build_csr_kernel(const long long* __restrict__ ei, int* __restrict__ cursor,
                                 int* __restrict__ srcidx) {
    int e = blockIdx.x * blockDim.x + threadIdx.x;
    if (e >= N_EDGES_C) return;
    int d = (int)ei[(long long)N_EDGES_C + e];
    int s = (int)ei[e];
    int pos = atomicAdd(&cursor[d], 1);
    srcidx[pos] = s;
}

// ================= GIN aggregation: agg[i] = x[i] + sum_{j->i} x[j] =================
// One wave per node, lane owns 4 channels. Indices are fetched 32-at-a-time by the
// wave (one vector load) and broadcast via lane shuffle, so all gather addresses in a
// batch are known after a single index load -> gathers stay pipelined, no atomics.
__global__ void aggregate_kernel(const float* __restrict__ x, const int* __restrict__ offs,
                                 const int* __restrict__ srcidx, float* __restrict__ agg) {
    int gid  = blockIdx.x * blockDim.x + threadIdx.x;
    int node = gid >> 5;
    if (node >= N_NODES_C) return;
    int lane = gid & 31;
    const size_t cb = (size_t)lane * 4;
    float4 acc = *(const float4*)(x + (size_t)node * DIM + cb);   // self term (eps = 0)
    const int beg = offs[node];
    const int end = offs[node + 1];
    for (int base = beg; base < end; base += 32) {
        const int n = min(32, end - base);
        int idx = 0;
        if (base + lane < end) idx = srcidx[base + lane];         // 1 load covers <=32 edges
        for (int k = 0; k < n; ++k) {
            const int s = __shfl(idx, k, 32);                     // VALU broadcast
            const float4 v = *(const float4*)(x + (size_t)s * DIM + cb);
            acc.x += v.x; acc.y += v.y; acc.z += v.z; acc.w += v.w;
        }
    }
    *(float4*)(agg + (size_t)node * DIM + cb) = acc;
}

// ---------------- pack a 128x128 f32 weight into WMMA B fragments ----------------
// B-frag element j of lane L (tile t, k-chunk kc): col = t*16 + (L&15), k = kc*32 + (L>>4)*8 + (j<8 ? j : j+8)
__global__ void pack_wfrag_kernel(const float* __restrict__ W, _Float16* __restrict__ out) {
    int idx  = blockIdx.x * blockDim.x + threadIdx.x;   // 16384 threads total
    int j    = idx & 15;
    int lane = (idx >> 4) & 31;
    int t    = (idx >> 9) & 7;
    int kc   = (idx >> 12) & 3;
    int col  = t * 16 + (lane & 15);
    int k    = kc * 32 + (lane >> 4) * 8 + (j < 8 ? j : j + 8);
    out[idx] = (_Float16)W[k * DIM + col];
}

// ---------------- WMMA GEMM: out[N x 128] = act(A[N x 128] @ W + b) ----------------
// One wave computes a 16x128 strip: 8 accumulator tiles, K=128 in 4 chunks of 32.
template<bool A_HALF, bool OUT_HALF, bool RELU>
__global__ __launch_bounds__(256) void gin_gemm_kernel(
        const void* __restrict__ Ain, const _Float16* __restrict__ Wfrag,
        const float* __restrict__ bias, void* __restrict__ Out, int nrows)
{
    const int lane    = threadIdx.x & 31;
    const int wave    = threadIdx.x >> 5;
    const int rowtile = blockIdx.x * (blockDim.x >> 5) + wave;
    if (rowtile * 16 >= nrows) return;              // wave-uniform guard (EXEC stays all-ones)
    const int rowbase = rowtile * 16;
    const int hs      = lane >> 4;
    const int l16     = lane & 15;
    const int arow    = rowbase + l16;

    v8f acc[8] = {};

    #pragma unroll
    for (int kc = 0; kc < 4; ++kc) {
        // A fragment: ISA 16-bit A 16x32 layout -> element j holds K = kc*32 + hs*8 + (j<8 ? j : j+8)
        v16h afrag;
        if (A_HALF) {
            const _Float16* ap = (const _Float16*)Ain + (size_t)arow * DIM + kc * 32 + hs * 8;
            #pragma unroll
            for (int j = 0; j < 8; ++j) afrag[j] = ap[j];
            #pragma unroll
            for (int j = 0; j < 8; ++j) afrag[8 + j] = ap[16 + j];
        } else {
            const float* ap = (const float*)Ain + (size_t)arow * DIM + kc * 32 + hs * 8;
            #pragma unroll
            for (int j = 0; j < 8; ++j) afrag[j] = (_Float16)ap[j];
            #pragma unroll
            for (int j = 0; j < 8; ++j) afrag[8 + j] = (_Float16)ap[16 + j];
        }
        #pragma unroll
        for (int t = 0; t < 8; ++t) {
            const v16h bfrag = *(const v16h*)(Wfrag + (((kc * 8 + t) * 32 + lane) << 4));
            acc[t] = __builtin_amdgcn_wmma_f32_16x16x32_f16(
                         false, afrag, false, bfrag, (short)0, acc[t], false, false);
        }
    }

    // C/D layout: col = lane&15, row = r + 8*hs
    #pragma unroll
    for (int t = 0; t < 8; ++t) {
        const int col  = t * 16 + l16;
        const float bv = bias[col];
        #pragma unroll
        for (int r = 0; r < 8; ++r) {
            float v = acc[t][r] + bv;
            if (RELU) v = fmaxf(v, 0.0f);
            const int m = rowbase + r + hs * 8;
            if (OUT_HALF) ((_Float16*)Out)[(size_t)m * DIM + col] = (_Float16)v;
            else          ((float*)Out)[(size_t)m * DIM + col]    = v;
        }
    }
}

// ---------------- pooled segment sum (batch is sorted): register-accumulate runs ----------------
__global__ void pool_kernel(const float* __restrict__ h, const long long* __restrict__ batch,
                            float* __restrict__ pool, float* __restrict__ cnt) {
    const int c    = threadIdx.x;       // 128 threads = channels
    const int base = blockIdx.x * 64;   // 64 nodes per block
    float acc = 0.0f, cacc = 0.0f;
    int curg = -1;
    for (int k = 0; k < 64; ++k) {
        int i = base + k;
        if (i >= N_NODES_C) break;
        int g = (int)batch[i];
        if (g != curg) {
            if (curg >= 0) {
                atomicAdd(&pool[curg * DIM + c], acc);
                if (c == 0) atomicAdd(&cnt[curg], cacc);
            }
            curg = g; acc = 0.0f; cacc = 0.0f;
        }
        acc  += h[(size_t)i * DIM + c];
        cacc += 1.0f;
    }
    if (curg >= 0) {
        atomicAdd(&pool[curg * DIM + c], acc);
        if (c == 0) atomicAdd(&cnt[curg], cacc);
    }
}

// ---------------- head MLP: 64 x (128 -> 256 -> 2), one block per graph ----------------
__global__ void mlp_head_kernel(const float* __restrict__ pool, const float* __restrict__ cnt,
                                const float* __restrict__ lw, const float* __restrict__ lb,
                                const float* __restrict__ l2w, const float* __restrict__ l2b,
                                float* __restrict__ out) {
    __shared__ float p[DIM];
    __shared__ float hid[POOLDIM];
    const int g = blockIdx.x;
    const int t = threadIdx.x;          // 256 threads
    if (t < DIM) p[t] = pool[g * DIM + t] / fmaxf(cnt[g], 1.0f);
    __syncthreads();
    float s = lb[t];
    #pragma unroll 4
    for (int k = 0; k < DIM; ++k) s += p[k] * lw[k * POOLDIM + t];
    hid[t] = fmaxf(s, 0.0f);
    __syncthreads();
    if (t < 2) {
        float o = l2b[t];
        for (int k = 0; k < POOLDIM; ++k) o += hid[k] * l2w[k * 2 + t];
        out[g * 2 + t] = o;
    }
}

extern "C" void kernel_launch(void* const* d_in, const int* in_sizes, int n_in,
                              void* d_out, int out_size, void* d_ws, size_t ws_size,
                              hipStream_t stream) {
    const float*     x     = (const float*)d_in[0];
    const long long* ei    = (const long long*)d_in[1];
    const long long* batch = (const long long*)d_in[2];
    const float* b1a = (const float*)d_in[5];
    const float* b1b = (const float*)d_in[7];
    const float* b2a = (const float*)d_in[9];
    const float* b2b = (const float*)d_in[11];
    const float* b3a = (const float*)d_in[13];
    const float* b3b = (const float*)d_in[15];
    const float* lin_w  = (const float*)d_in[16];
    const float* lin_b  = (const float*)d_in[17];
    const float* lin2_w = (const float*)d_in[18];
    const float* lin2_b = (const float*)d_in[19];

    char* ws = (char*)d_ws;
    float*     FA     = (float*)(ws + FA_OFF);
    float*     FB     = (float*)(ws + FB_OFF);
    _Float16*  TMP    = (_Float16*)(ws + TMP_OFF);
    _Float16*  WH     = (_Float16*)(ws + WH_OFF);
    float*     POOL   = (float*)(ws + POOL_OFF);
    float*     CNT    = POOL + NGRAPH * DIM;
    int*       OFFS   = (int*)(ws + OFFS_OFF);
    int*       CURS   = (int*)(ws + CURS_OFF);
    int*       SRCIDX = (int*)(ws + SRC_OFF);
    int*       BSUM   = (int*)(ws + BSUM_OFF);

    const int gemmBlocks = ((N_NODES_C / 16) + 7) / 8;       // 782
    const int edgeBlocks = (N_EDGES_C + 255) / 256;          // 6250
    const int nodeBlocks = (N_NODES_C + 255) / 256;          // 391
    const int aggBlocks  = (N_NODES_C * 32 + 255) / 256;     // 12500
    const int poolBlocks = (N_NODES_C + 63) / 64;            // 1563

    // ---- build CSR (dst -> list of src), reused by all three layers
    zero_u32_kernel<<<nodeBlocks, 256, 0, stream>>>((unsigned int*)CURS, N_NODES_C);
    degree_kernel<<<edgeBlocks, 256, 0, stream>>>(ei, CURS);
    scan_partial_kernel<<<nodeBlocks, 256, 0, stream>>>(CURS, OFFS, BSUM);
    scan_bsum_kernel<<<1, 32, 0, stream>>>(BSUM, nodeBlocks);
    scan_addback_kernel<<<nodeBlocks, 256, 0, stream>>>(OFFS, BSUM);
    hipMemcpyAsync(CURS, OFFS, (size_t)N_NODES_C * sizeof(int), hipMemcpyDeviceToDevice, stream);
    build_csr_kernel<<<edgeBlocks, 256, 0, stream>>>(ei, CURS, SRCIDX);

    // ---- pack weights into WMMA B-fragment order (ws not preserved across calls)
    pack_wfrag_kernel<<<64, 256, 0, stream>>>((const float*)d_in[4],  WH + 0 * 16384);
    pack_wfrag_kernel<<<64, 256, 0, stream>>>((const float*)d_in[6],  WH + 1 * 16384);
    pack_wfrag_kernel<<<64, 256, 0, stream>>>((const float*)d_in[8],  WH + 2 * 16384);
    pack_wfrag_kernel<<<64, 256, 0, stream>>>((const float*)d_in[10], WH + 3 * 16384);
    pack_wfrag_kernel<<<64, 256, 0, stream>>>((const float*)d_in[12], WH + 4 * 16384);
    pack_wfrag_kernel<<<64, 256, 0, stream>>>((const float*)d_in[14], WH + 5 * 16384);
    zero_u32_kernel<<<(NGRAPH * DIM + NGRAPH + 255) / 256, 256, 0, stream>>>(
        (unsigned int*)POOL, NGRAPH * DIM + NGRAPH);

    // ---- layer 1: h1 = relu( mlp1( x + sum_j x_j ) ) -> FB
    aggregate_kernel<<<aggBlocks, 256, 0, stream>>>(x, OFFS, SRCIDX, FA);
    gin_gemm_kernel<false, true,  true ><<<gemmBlocks, 256, 0, stream>>>(FA, WH + 0 * 16384, b1a, TMP, N_NODES_C);
    gin_gemm_kernel<true,  false, true ><<<gemmBlocks, 256, 0, stream>>>(TMP, WH + 1 * 16384, b1b, FB, N_NODES_C);

    // ---- layer 2: h2 = relu( mlp2( h1 + sum_j h1_j ) ) -> FB
    aggregate_kernel<<<aggBlocks, 256, 0, stream>>>(FB, OFFS, SRCIDX, FA);
    gin_gemm_kernel<false, true,  true ><<<gemmBlocks, 256, 0, stream>>>(FA, WH + 2 * 16384, b2a, TMP, N_NODES_C);
    gin_gemm_kernel<true,  false, true ><<<gemmBlocks, 256, 0, stream>>>(TMP, WH + 3 * 16384, b2b, FB, N_NODES_C);

    // ---- layer 3: h3 = mlp3( h2 + sum_j h2_j )  (no outer relu) -> FB
    aggregate_kernel<<<aggBlocks, 256, 0, stream>>>(FB, OFFS, SRCIDX, FA);
    gin_gemm_kernel<false, true,  true ><<<gemmBlocks, 256, 0, stream>>>(FA, WH + 4 * 16384, b3a, TMP, N_NODES_C);
    gin_gemm_kernel<true,  false, false><<<gemmBlocks, 256, 0, stream>>>(TMP, WH + 5 * 16384, b3b, FB, N_NODES_C);

    // ---- mean pool + head MLP
    pool_kernel<<<poolBlocks, 128, 0, stream>>>(FB, batch, POOL, CNT);
    mlp_head_kernel<<<NGRAPH, 256, 0, stream>>>(POOL, CNT, lin_w, lin_b, lin2_w, lin2_b, (float*)d_out);
}
